// LogLinearDeltaProductBlock_15358803050897
// MI455X (gfx1250) — matrix operand order repaired and checked by reading, not compiled, not run on hardware
//
#include <hip/hip_runtime.h>
#include <hip/hip_bf16.h>
#include <cstdint>

// ---------------------------------------------------------------------------
// LogLinearDeltaProduct block for MI455X (gfx1250).
//  - All dense GEMMs use v_wmma_f32_16x16x32_bf16, fused activation epilogues.
//  - B tiles staged in LDS via global_load_async_to_lds_b128 (ASYNCcnt),
//    double-buffered with a peeled epilogue so the steady-state K-loop is
//    branch-free: issue(next) -> s_wait_asynccnt 2 -> barrier -> 8x ds_load ->
//    4x wmma -> barrier.
//  - Sequential gated DeltaProduct recurrence: one workgroup per head, 64x64
//    f32 state in LDS; per-step k/v/q vectors fetched global->LDS via
//    global_load_async_to_lds_b32.
// ---------------------------------------------------------------------------

typedef __bf16 bf16;
typedef __attribute__((ext_vector_type(16))) __bf16 v16bf;
typedef __attribute__((ext_vector_type(8)))  __bf16 v8bf;
typedef __attribute__((ext_vector_type(8)))  float  v8f;

#define T_     2048
#define D_     1024
#define H_     16
#define K_     64
#define V_     64
#define NH_    3
#define C_     64
#define LMAX_  16
#define NCHUNK 32
#define BSTRIDE 40   // bf16 elements per staged B column (32 data + 8 pad)

// ---------------------------------------------------------------- converts --
__global__ __launch_bounds__(256) void f32_to_bf16_kernel(
    const float* __restrict__ src, bf16* __restrict__ dst, int n) {
  int i = blockIdx.x * 256 + threadIdx.x;
  if (i < n) dst[i] = (bf16)src[i];
}

// dst[n*Kd + kd] = src[kd*N + n]   (weight -> transposed bf16 for WMMA B-frags)
__global__ __launch_bounds__(256) void transpose_f32_to_bf16_kernel(
    const float* __restrict__ src, bf16* __restrict__ dst, int Kd, int N) {
  int i = blockIdx.x * 256 + threadIdx.x;
  if (i < Kd * N) {
    int n = i / Kd, kd = i % Kd;
    dst[i] = (bf16)src[(size_t)kd * N + n];
  }
}

// ------------------------------------------------------------------- GEMM ---
// modes: 0=none 1=silu 2=sigmoid 3=log_sigmoid 4=softplus(colscale[col]*x)
template <int MODE>
__device__ __forceinline__ float gemm_act(float x, float s) {
  if (MODE == 1) return x / (1.f + __expf(-x));
  if (MODE == 2) return 1.f / (1.f + __expf(-x));
  if (MODE == 3) {
    float z = -x;
    float sp = (z > 20.f) ? z : __logf(1.f + __expf(z));
    return -sp;
  }
  if (MODE == 4) {
    float z = s * x;
    return (z > 20.f) ? z : __logf(1.f + __expf(z));
  }
  return x;
}

// C[M,N] f32 = A[M,Kd] bf16 row-major x Bt[N,Kd] bf16 (B pre-transposed).
// Block = 128 threads = 4 waves; tile 64x64; wave w -> 16-row M strip.
// B tile (64 cols x 32 k) is identical for all 4 waves -> staged in LDS via
// async DMA, double-buffered with peeled epilogue.
template <int NSUB, int MODE>
__global__ __launch_bounds__(128) void wmma_gemm_kernel(
    const bf16* __restrict__ A, const bf16* __restrict__ Bt,
    float* __restrict__ C, int M, int N, int Kd,
    const float* __restrict__ colscale) {
  __shared__ bf16 sB[2][64 * BSTRIDE];

  const int tid   = threadIdx.x;
  const int lane  = tid & 31;
  const int wave  = tid >> 5;
  const int tileM = blockIdx.y * 64 + wave * 16;
  const int tileN = blockIdx.x * 64;

  const int l15    = lane & 15;
  const int hi     = lane >> 4;   // 0: K-half {0-7,16-23} / 1: {8-15,24-31}
  const int rowA   = tileM + l15;
  const int khalfA = hi * 8;
  const int koffB  = hi * 8;

  // async B staging: thread -> (col = tid>>1, half = tid&1), 32 bytes/slot
  const int bcol  = tid >> 1;
  const int bhalf = tid & 1;
  const bool bact = (bcol < NSUB * 16);
  const bf16* bsrc = Bt + (size_t)(tileN + bcol) * Kd + bhalf * 16;
  const uint32_t lds0 =
      (uint32_t)(uintptr_t)(&sB[0][bcol * BSTRIDE + bhalf * 16]);
  const uint32_t lds1 =
      (uint32_t)(uintptr_t)(&sB[1][bcol * BSTRIDE + bhalf * 16]);

  auto issueB = [&](int kk, int buf) {
    if (!bact) return;
    uint64_t ga = (uint64_t)(uintptr_t)(bsrc + kk);
    uint32_t la = buf ? lds1 : lds0;
    // INST_OFFSET applies to both LDS and global address -> contiguous 32B.
    asm volatile("global_load_async_to_lds_b128 %0, %1, off"
                 :: "v"(la), "v"(ga) : "memory");
    asm volatile("global_load_async_to_lds_b128 %0, %1, off offset:16"
                 :: "v"(la), "v"(ga) : "memory");
  };

  v8f acc[NSUB];
#pragma unroll
  for (int s = 0; s < NSUB; ++s) acc[s] = (v8f)0.f;

  // Load all B fragments first so the ds_loads clause together and a single
  // wait covers them, then issue the WMMAs back-to-back.
  auto compute = [&](int kk, int buf) {
    const bf16* ap = A + (size_t)rowA * Kd + kk + khalfA;
    v8bf a0 = *(const v8bf*)ap;
    v8bf a1 = *(const v8bf*)(ap + 16);
    v16bf af = __builtin_shufflevector(a0, a1,
        0,1,2,3,4,5,6,7,8,9,10,11,12,13,14,15);

    const bf16* sbuf = sB[buf];
    v16bf bfv[NSUB];
#pragma unroll
    for (int s = 0; s < NSUB; ++s) {
      const bf16* bp = sbuf + (s * 16 + l15) * BSTRIDE + koffB;
      v8bf b0 = *(const v8bf*)bp;
      v8bf b1 = *(const v8bf*)(bp + 16);
      bfv[s] = __builtin_shufflevector(b0, b1,
          0,1,2,3,4,5,6,7,8,9,10,11,12,13,14,15);
    }
#pragma unroll
    for (int s = 0; s < NSUB; ++s) {
      acc[s] = __builtin_amdgcn_wmma_f32_16x16x32_bf16(
          false, af, false, bfv[s], (short)0, acc[s], false, false);
    }
  };

  issueB(0, 0);
  int kk = 0, buf = 0;
  // steady state: branch-free body
  for (; kk + 32 < Kd; kk += 32, buf ^= 1) {
    issueB(kk + 32, buf ^ 1);
    asm volatile("s_wait_asynccnt 0x2" ::: "memory");
    __syncthreads();           // current buffer visible to all waves
    __builtin_prefetch((const void*)(A + (size_t)rowA * Kd + kk + 32 + khalfA), 0, 3);
    compute(kk, buf);
    __syncthreads();           // don't let a fast wave overwrite buf^1 early
  }
  // peeled last iteration
  asm volatile("s_wait_asynccnt 0x0" ::: "memory");
  __syncthreads();
  compute(kk, buf);

  // C layout: VGPR i -> row = tileM + hi*8 + i, col = subtileBase + l15.
  const int rbase = tileM + hi * 8;
#pragma unroll
  for (int s = 0; s < NSUB; ++s) {
    const int col = tileN + s * 16 + l15;
    const float cs = (MODE == 4) ? colscale[col] : 0.f;
#pragma unroll
    for (int i = 0; i < 8; ++i) {
      C[(size_t)(rbase + i) * N + col] = gemm_act<MODE>(acc[s][i], cs);
    }
  }
}

// ------------------------------------------------------------- chunk means --
__global__ __launch_bounds__(256) void chunk_mean_kernel(
    const float* __restrict__ lw, float* __restrict__ lwc) {
  int ci = blockIdx.x, c = threadIdx.x;
  float s = 0.f;
  for (int t = 0; t < C_; ++t) s += lw[(size_t)(ci * C_ + t) * 256 + c];
  lwc[ci * 256 + c] = s * (1.f / (float)C_);
}

// -------------------------------------------------------------- recurrence --
// One workgroup per head. State S (64x64 f32) in LDS. Binary-counter tree of
// LMAX states per head in global ws; occupied levels at chunk ci = bits of ci.
__global__ __launch_bounds__(256) void delta_recurrence_kernel(
    const float* __restrict__ qb,    // (T,H,K)
    const float* __restrict__ kb,    // (T,H,NH,K)  silu applied
    const float* __restrict__ vb,    // (T,H,NH,V)  silu applied
    const float* __restrict__ gb,    // (T,H)       log-sigmoid applied
    const float* __restrict__ betab, // (T,H,NH)    sigmoid applied
    const float* __restrict__ lwc,   // (NCHUNK,H,LMAX)
    float* __restrict__ yb,          // (T,H,V)
    float* __restrict__ states) {    // (LMAX,H,K*V)
  const int h   = blockIdx.x;
  const int tid = threadIdx.x;
  const int p   = tid >> 6;    // k-quarter 0..3
  const int vv  = tid & 63;    // v index

  __shared__ float sS[K_ * V_];
  __shared__ float sPart[256];
  __shared__ float sK[K_];
  __shared__ float sV[V_];
  __shared__ float sD[V_];

  const uint32_t ldsK = (uint32_t)(uintptr_t)(&sK[tid & 63]);
  const uint32_t ldsV = (uint32_t)(uintptr_t)(&sV[tid & 63]);
  const uint32_t ldsD = (uint32_t)(uintptr_t)(&sD[tid & 63]);

  for (int ci = 0; ci < NCHUNK; ++ci) {
    // init: S = sum over occupied levels (bits of ci) of lwc * state[l]
#pragma unroll 4
    for (int i = 0; i < 16; ++i) {
      int flat = tid + 256 * i;
      float acc = 0.f;
      for (int l = 0; l < LMAX_; ++l) {
        if ((ci >> l) & 1)
          acc += lwc[(ci * H_ + h) * LMAX_ + l] *
                 states[((size_t)(l * H_ + h)) * (K_ * V_) + flat];
      }
      sS[flat] = acc;
    }
    __syncthreads();

    for (int t = 0; t < C_; ++t) {
      const int tt = ci * C_ + t;
      // gated decay
      const float decay = __expf(gb[tt * H_ + h]);
#pragma unroll 4
      for (int i = 0; i < 16; ++i) sS[tid + 256 * i] *= decay;
      __syncthreads();

      // NH Householder (delta-rule) updates
      for (int j = 0; j < NH_; ++j) {
        if (tid < K_) {
          uint64_t gk = (uint64_t)(uintptr_t)(
              kb + ((size_t)(tt * H_ + h) * NH_ + j) * K_ + tid);
          uint64_t gv = (uint64_t)(uintptr_t)(
              vb + ((size_t)(tt * H_ + h) * NH_ + j) * V_ + tid);
          asm volatile("global_load_async_to_lds_b32 %0, %1, off"
                       :: "v"(ldsK), "v"(gk) : "memory");
          asm volatile("global_load_async_to_lds_b32 %0, %1, off"
                       :: "v"(ldsV), "v"(gv) : "memory");
        }
        asm volatile("s_wait_asynccnt 0x0" ::: "memory");
        __syncthreads();
        float part = 0.f;
#pragma unroll
        for (int kk = 0; kk < 16; ++kk)
          part += sK[p * 16 + kk] * sS[(p * 16 + kk) * V_ + vv];
        sPart[tid] = part;
        __syncthreads();
        if (tid < V_) {
          float ks = sPart[tid] + sPart[64 + tid] + sPart[128 + tid] + sPart[192 + tid];
          float beta = betab[tt * (H_ * NH_) + h * NH_ + j];
          sD[tid] = beta * (sV[tid] - ks);
        }
        __syncthreads();
#pragma unroll 4
        for (int i = 0; i < 16; ++i) {
          int flat = tid + 256 * i;
          sS[flat] += sK[flat >> 6] * sD[flat & 63];
        }
        __syncthreads();
      }

      // output o = scale * (q . S)   (scale folded into the final write)
      if (tid < K_) {
        uint64_t gq = (uint64_t)(uintptr_t)(
            qb + (size_t)tt * (H_ * K_) + h * K_ + tid);
        asm volatile("global_load_async_to_lds_b32 %0, %1, off"
                     :: "v"(ldsD), "v"(gq) : "memory");
      }
      asm volatile("s_wait_asynccnt 0x0" ::: "memory");
      __syncthreads();
      float part = 0.f;
#pragma unroll
      for (int kk = 0; kk < 16; ++kk)
        part += sD[p * 16 + kk] * sS[(p * 16 + kk) * V_ + vv];
      sPart[tid] = part;
      __syncthreads();
      if (tid < V_)
        yb[(size_t)tt * (H_ * V_) + h * V_ + tid] =
            (sPart[tid] + sPart[64 + tid] + sPart[128 + tid] + sPart[192 + tid])
            * 0.125f;
      __syncthreads();
    }

    // tree merge: tz = ctz(ci+1); levels j<tz are all occupied (trailing ones)
    const int tz = __ffs(ci + 1) - 1;
#pragma unroll 4
    for (int i = 0; i < 16; ++i) {
      int flat = tid + 256 * i;
      float val = sS[flat];
      for (int j2 = 0; j2 < tz; ++j2)
        val += states[((size_t)(j2 * H_ + h)) * (K_ * V_) + flat];
      states[((size_t)(tz * H_ + h)) * (K_ * V_) + flat] = val;
    }
    __syncthreads();
  }
}

// --------------------------------------------------------------- groupnorm --
__global__ __launch_bounds__(256) void gn_stats_kernel(
    const float* __restrict__ y, float* __restrict__ stats) {
  const int h = blockIdx.x, tid = threadIdx.x;
  float s = 0.f, s2 = 0.f;
  for (int flat = tid; flat < T_ * V_; flat += 256) {
    int t = flat >> 6, v = flat & 63;
    float val = y[(size_t)t * (H_ * V_) + h * V_ + v];
    s += val; s2 += val * val;
  }
  __shared__ float rs[256], rs2[256];
  rs[tid] = s; rs2[tid] = s2;
  __syncthreads();
  for (int off = 128; off > 0; off >>= 1) {
    if (tid < off) { rs[tid] += rs[tid + off]; rs2[tid] += rs2[tid + off]; }
    __syncthreads();
  }
  if (tid == 0) {
    const float inv = 1.f / (float)(T_ * V_);
    float mean = rs[0] * inv;
    float var  = rs2[0] * inv - mean * mean;
    stats[h * 2]     = mean;
    stats[h * 2 + 1] = rsqrtf(var + 1e-5f);
  }
}

__global__ __launch_bounds__(256) void gn_apply_kernel(
    const float* __restrict__ y, const float* __restrict__ stats,
    const float* __restrict__ gnw, const float* __restrict__ gnb,
    const float* __restrict__ sog, bf16* __restrict__ ygb) {
  int idx = blockIdx.x * 256 + threadIdx.x;
  if (idx < T_ * (H_ * V_)) {
    int c = idx & (H_ * V_ - 1);
    int h = c >> 6;
    float val = (y[idx] - stats[h * 2]) * stats[h * 2 + 1] * gnw[c] + gnb[c];
    ygb[idx] = (bf16)(val * sog[idx]);
  }
}

// ------------------------------------------------------------------ launch --
template <int NSUB, int MODE>
static void launch_gemm(const bf16* A, const bf16* Bt, float* C,
                        int M, int N, int Kd, const float* cs, hipStream_t s) {
  dim3 grid((unsigned)((N + 63) / 64), (unsigned)(M / 64));
  wmma_gemm_kernel<NSUB, MODE><<<grid, 128, 0, s>>>(A, Bt, C, M, N, Kd, cs);
}

extern "C" void kernel_launch(void* const* d_in, const int* in_sizes, int n_in,
                              void* d_out, int out_size, void* d_ws, size_t ws_size,
                              hipStream_t stream) {
  const float* x   = (const float*)d_in[0];
  // d_in[1]=cos, d_in[2]=sin are unused by the reference
  const float* Wq  = (const float*)d_in[3];
  const float* Wk  = (const float*)d_in[4];
  const float* Wv  = (const float*)d_in[5];
  const float* Wb  = (const float*)d_in[6];
  const float* Wa  = (const float*)d_in[7];
  const float* Wg  = (const float*)d_in[8];
  const float* Wo  = (const float*)d_in[9];
  const float* Wl  = (const float*)d_in[10];
  const float* Lm  = (const float*)d_in[11];
  const float* gnw = (const float*)d_in[12];
  const float* gnb = (const float*)d_in[13];
  float* out = (float*)d_out;

  char* ws = (char*)d_ws;
  size_t off = 0;
  auto alloc = [&](size_t bytes) -> char* {
    off = (off + 255) & ~(size_t)255;
    char* p = ws + off;
    off += bytes;
    return p;
  };

  bf16* xb  = (bf16*)alloc((size_t)T_ * D_ * 2);
  bf16* WqT = (bf16*)alloc((size_t)1024 * D_ * 2);
  bf16* WkT = (bf16*)alloc((size_t)3072 * D_ * 2);
  bf16* WvT = (bf16*)alloc((size_t)3072 * D_ * 2);
  bf16* WbT = (bf16*)alloc((size_t)48   * D_ * 2);
  bf16* WaT = (bf16*)alloc((size_t)16   * D_ * 2);
  bf16* WgT = (bf16*)alloc((size_t)1024 * D_ * 2);
  bf16* WlT = (bf16*)alloc((size_t)256  * D_ * 2);
  bf16* WoT = (bf16*)alloc((size_t)1024 * D_ * 2);

  float* qbuf   = (float*)alloc((size_t)T_ * 1024 * 4);
  float* kbuf   = (float*)alloc((size_t)T_ * 3072 * 4);
  float* vbuf   = (float*)alloc((size_t)T_ * 3072 * 4);
  float* betab  = (float*)alloc((size_t)T_ * 48 * 4);
  float* gbuf   = (float*)alloc((size_t)T_ * 16 * 4);
  float* sog    = (float*)alloc((size_t)T_ * 1024 * 4);
  float* lwbuf  = (float*)alloc((size_t)T_ * 256 * 4);
  float* lwc    = (float*)alloc((size_t)NCHUNK * 256 * 4);
  float* ybuf   = (float*)alloc((size_t)T_ * 1024 * 4);
  float* states = (float*)alloc((size_t)LMAX_ * H_ * K_ * V_ * 4);
  float* stats  = (float*)alloc((size_t)H_ * 2 * 4);
  bf16*  ygb    = (bf16*)alloc((size_t)T_ * 1024 * 2);

  // --- bf16 conversions / weight transposes
  {
    int n = T_ * D_;
    f32_to_bf16_kernel<<<(n + 255) / 256, 256, 0, stream>>>(x, xb, n);
  }
  auto tr = [&](const float* src, bf16* dst, int N) {
    int n = D_ * N;
    transpose_f32_to_bf16_kernel<<<(n + 255) / 256, 256, 0, stream>>>(src, dst, D_, N);
  };
  tr(Wq, WqT, 1024); tr(Wk, WkT, 3072); tr(Wv, WvT, 3072); tr(Wb, WbT, 48);
  tr(Wa, WaT, 16);   tr(Wg, WgT, 1024); tr(Wl, WlT, 256);  tr(Wo, WoT, 1024);

  // --- projection GEMMs with fused activations
  launch_gemm<4, 0>(xb, WqT, qbuf,  T_, 1024, D_, nullptr, stream);
  launch_gemm<4, 1>(xb, WkT, kbuf,  T_, 3072, D_, nullptr, stream);
  launch_gemm<4, 1>(xb, WvT, vbuf,  T_, 3072, D_, nullptr, stream);
  launch_gemm<3, 2>(xb, WbT, betab, T_, 48,   D_, nullptr, stream);
  launch_gemm<1, 3>(xb, WaT, gbuf,  T_, 16,   D_, nullptr, stream);
  launch_gemm<4, 2>(xb, WgT, sog,   T_, 1024, D_, nullptr, stream);
  launch_gemm<4, 4>(xb, WlT, lwbuf, T_, 256,  D_, Lm,      stream);

  // --- chunk-mean of level weights
  chunk_mean_kernel<<<NCHUNK, 256, 0, stream>>>(lwbuf, lwc);

  // --- sequential gated DeltaProduct recurrence (one WGP per head)
  delta_recurrence_kernel<<<H_, 256, 0, stream>>>(qbuf, kbuf, vbuf, gbuf, betab,
                                                  lwc, ybuf, states);

  // --- groupnorm over (T,V) per head, gate, convert to bf16
  gn_stats_kernel<<<H_, 256, 0, stream>>>(ybuf, stats);
  {
    int n = T_ * 1024;
    gn_apply_kernel<<<(n + 255) / 256, 256, 0, stream>>>(ybuf, stats, gnw, gnb, sog, ygb);
  }

  // --- output projection
  launch_gemm<4, 0>(ygb, WoT, out, T_, 1024, D_, nullptr, stream);
}
